// EntityEncoder_35227321762253
// MI455X (gfx1250) — matrix operand order repaired and checked by reading, compiled, not run on hardware
//
#include <hip/hip_runtime.h>
#include <math.h>

typedef __attribute__((ext_vector_type(16))) _Float16    v16h;
typedef __attribute__((ext_vector_type(8)))  float       v8f;
typedef __attribute__((ext_vector_type(4)))  unsigned int u32x4;
typedef __attribute__((ext_vector_type(2)))  unsigned int u32x2;

#define VOCAB 100000
#define EMBD  128
#define H1    256
#define H2    128
#define SEQ   50
#define MPAD  64   // 50 tokens padded to 4 WMMA M-tiles

union Frag16 { v16h v; u32x4 q[2]; };
union Pack4h { _Float16 h[4]; u32x2 u; };

__device__ __forceinline__ float wave_sum32(float v) {
#pragma unroll
    for (int m = 16; m >= 1; m >>= 1) v += __shfl_xor(v, m, 32);
    return v;
}

// Branchless tanh-form GELU: 0.5x(1+tanh(0.7978846(x+0.044715x^3))) = x*sigmoid(u)
// with u = 1.5957691(x + 0.044715 x^3). Uses v_exp_f32 + v_rcp_f32, no branches.
__device__ __forceinline__ float gelu_fast(float x) {
    float u = x * (1.5957691f + 0.07135481f * x * x);
    u = fminf(fmaxf(u, -40.0f), 40.0f);
    float e = __builtin_amdgcn_exp2f(u * -1.4426950408889634f);
    return x * __builtin_amdgcn_rcpf(1.0f + e);
}

// CDNA5 async global->LDS copy (16B per lane), tracked by ASYNCcnt.
__device__ __forceinline__ void async_copy16(unsigned lds_off, const void* gaddr) {
    asm volatile("global_load_async_to_lds_b128 %0, %1, off"
                 :: "v"(lds_off), "v"(gaddr) : "memory");
}
__device__ __forceinline__ void async_wait0() {
    asm volatile("s_wait_asynccnt 0x0" ::: "memory");
}

// ---------------------------------------------------------------------------
// Prep: convert w1[128][256], w2[256][128] (f32, K x N row-major) into
// K-contiguous f16 B-fragment layouts w1t[N][K], w2t[N][K] in workspace.
// ---------------------------------------------------------------------------
__global__ void prep_weights_f16(const float* __restrict__ w1,
                                 const float* __restrict__ w2,
                                 _Float16* __restrict__ w1t,
                                 _Float16* __restrict__ w2t) {
    int idx = blockIdx.x * blockDim.x + threadIdx.x;
    if (idx < EMBD * H1) {                  // w1t[n][k] = w1[k][n]
        int n = idx / EMBD, k = idx % EMBD;
        w1t[n * EMBD + k] = (_Float16)w1[k * H1 + n];
    } else if (idx < EMBD * H1 + H1 * H2) { // w2t[n][k] = w2[k][n]
        int j = idx - EMBD * H1;
        int n = j / H1, k = j % H1;
        w2t[n * H1 + k] = (_Float16)w2[k * H2 + n];
    }
}

// ---------------------------------------------------------------------------
// Fused per-entity encoder: one block (8 wave32s) per entity.
// LDS: xA 64x128 f16 (16KB) | hA 64x256 f16 (32KB) | pb 64x128 f32 (32KB,
//      shared with the async gather staging xF) | pooled/stats/dotb
// ---------------------------------------------------------------------------
__global__ void __launch_bounds__(256)
entity_encoder_fused(const int*   __restrict__ ids,
                     const int*   __restrict__ lens,
                     const float* __restrict__ emb,
                     const float* __restrict__ ln1g, const float* __restrict__ ln1b,
                     const float* __restrict__ b1v,
                     const float* __restrict__ b2v,
                     const float* __restrict__ ln2g, const float* __restrict__ ln2b,
                     const float* __restrict__ fcw,  const float* __restrict__ fcb,
                     const _Float16* __restrict__ w1t,
                     const _Float16* __restrict__ w2t,
                     float* __restrict__ out) {
    extern __shared__ char smem[];
    _Float16* xA     = (_Float16*)(smem);
    _Float16* hA     = (_Float16*)(smem + 16384);
    float*    pb     = (float*)   (smem + 16384 + 32768);   // also gather staging xF
    float*    pooled = (float*)   (smem + 81920);
    float*    stats  = (float*)   (smem + 81920 + 512);
    float*    dotb   = (float*)   (smem + 81920 + 512 + 32);

    const int b     = blockIdx.x;
    const int tid   = threadIdx.x;
    const int lane  = tid & 31;
    const int wave  = tid >> 5;
    const int col15 = lane & 15;
    const int hiK   = (lane >= 16) ? 1 : 0;

    // ---- Stage 0: async gather of 50 embedding rows (f32) into LDS xF ----
    float* xF = pb;  // 64x128 f32 region, free until stage 3
    {
        const unsigned xF_base = (unsigned)(uintptr_t)(void*)xF;
        for (int c = tid; c < SEQ * 32; c += 256) {      // 32 x 16B chunks per row
            const int r   = c >> 5;
            const int off = (c & 31) << 4;               // byte offset in row
            const int id  = ids[b * SEQ + r];
            const char* g = (const char*)(emb + (size_t)id * EMBD) + off;
            async_copy16(xF_base + r * (EMBD * 4) + off, g);
        }
        async_wait0();
    }
    __syncthreads();

    // ---- Stage 1: LayerNorm1 (from LDS) -> xA (f16) -----------------------
    {
        const int c = lane * 4;
        const float g0 = ln1g[c], g1 = ln1g[c+1], g2 = ln1g[c+2], g3 = ln1g[c+3];
        const float e0 = ln1b[c], e1 = ln1b[c+1], e2 = ln1b[c+2], e3 = ln1b[c+3];
        for (int r = wave; r < MPAD; r += 8) {
            float x0 = 0.f, x1 = 0.f, x2 = 0.f, x3 = 0.f;
            if (r < SEQ) {
                const float* row = xF + r * EMBD;
                x0 = row[c]; x1 = row[c+1]; x2 = row[c+2]; x3 = row[c+3];
                const float s  = wave_sum32(x0 + x1 + x2 + x3);
                const float s2 = wave_sum32(x0*x0 + x1*x1 + x2*x2 + x3*x3);
                const float mu  = s  * (1.0f / EMBD);
                const float var = s2 * (1.0f / EMBD) - mu * mu;
                const float rs  = rsqrtf(var + 1e-12f);
                x0 = (x0 - mu) * rs * g0 + e0;
                x1 = (x1 - mu) * rs * g1 + e1;
                x2 = (x2 - mu) * rs * g2 + e2;
                x3 = (x3 - mu) * rs * g3 + e3;
            }
            Pack4h p;
            p.h[0] = (_Float16)x0; p.h[1] = (_Float16)x1;
            p.h[2] = (_Float16)x2; p.h[3] = (_Float16)x3;
            *(u32x2*)(xA + r * EMBD + c) = p.u;          // one ds_store_b64
        }
    }
    __syncthreads();

    // ---- Stage 2: GEMM1 (64x128 @ 128x256) + bias + GELU -> hA ------------
    // Wave owns 2 N-tiles; B fragments hoisted once, swept over 4 M-tiles.
#pragma unroll
    for (int nn = 0; nn < 2; ++nn) {
        const int nt = wave * 2 + nn;
        const int n  = nt * 16 + col15;                  // global H1 column
        Frag16 B4[4];
#pragma unroll
        for (int kc = 0; kc < 4; ++kc) {
            const u32x4* bp = (const u32x4*)(w1t + n * EMBD + kc * 32 + hiK * 16);
            B4[kc].q[0] = bp[0];
            B4[kc].q[1] = bp[1];
        }
        const float bias = b1v[n];
        for (int mt = 0; mt < 4; ++mt) {
            const int m = mt * 16 + col15;
            v8f acc = {};
#pragma unroll
            for (int kc = 0; kc < 4; ++kc) {
                Frag16 a;
                const u32x4* ap = (const u32x4*)(xA + m * EMBD + kc * 32 + hiK * 16);
                a.q[0] = ap[0];
                a.q[1] = ap[1];
                acc = __builtin_amdgcn_wmma_f32_16x16x32_f16(
                          false, a.v, false, B4[kc].v, (short)0, acc, false, false);
            }
#pragma unroll
            for (int v = 0; v < 8; ++v) {
                const float g = gelu_fast(acc[v] + bias);
                hA[(mt * 16 + v + hiK * 8) * H1 + n] = (_Float16)g;
            }
        }
    }
    __syncthreads();

    // ---- Stage 3: GEMM2 (64x256 @ 256x128) + bias -> pb (f32) -------------
    // Wave owns 1 N-tile; 8 B fragments hoisted, swept over 4 M-tiles.
    {
        const int nt = wave;
        const int n  = nt * 16 + col15;                  // global H2 column
        Frag16 B8[8];
#pragma unroll
        for (int kc = 0; kc < 8; ++kc) {
            const u32x4* bp = (const u32x4*)(w2t + n * H1 + kc * 32 + hiK * 16);
            B8[kc].q[0] = bp[0];
            B8[kc].q[1] = bp[1];
        }
        const float bias = b2v[n];
        for (int mt = 0; mt < 4; ++mt) {
            const int m = mt * 16 + col15;
            v8f acc = {};
#pragma unroll
            for (int kc = 0; kc < 8; ++kc) {
                Frag16 a;
                const u32x4* ap = (const u32x4*)(hA + m * H1 + kc * 32 + hiK * 16);
                a.q[0] = ap[0];
                a.q[1] = ap[1];
                acc = __builtin_amdgcn_wmma_f32_16x16x32_f16(
                          false, a.v, false, B8[kc].v, (short)0, acc, false, false);
            }
#pragma unroll
            for (int v = 0; v < 8; ++v) {
                pb[(mt * 16 + v + hiK * 8) * H2 + n] = acc[v] + bias;
            }
        }
    }
    __syncthreads();

    // ---- Stage 4: ragged max-pool over valid tokens -----------------------
    int len = lens[b];
    len = (len < 1) ? 1 : ((len > SEQ) ? SEQ : len);
    if (tid < H2) {
        float mx = -INFINITY;
        for (int r = 0; r < len; ++r) mx = fmaxf(mx, pb[r * H2 + tid]);
        pooled[tid] = mx;
    }
    __syncthreads();

    // ---- Stage 5: LayerNorm2 stats (wave 0) -------------------------------
    if (wave == 0) {
        const float a0 = pooled[lane],      a1 = pooled[lane + 32];
        const float a2 = pooled[lane + 64], a3 = pooled[lane + 96];
        const float s  = wave_sum32(a0 + a1 + a2 + a3);
        const float s2 = wave_sum32(a0*a0 + a1*a1 + a2*a2 + a3*a3);
        if (lane == 0) {
            const float mu  = s  * (1.0f / H2);
            const float var = s2 * (1.0f / H2) - mu * mu;
            stats[0] = mu;
            stats[1] = rsqrtf(var + 1e-12f);
        }
    }
    __syncthreads();

    // ---- Stage 6: normalize, scale, dot with fc_w -------------------------
    if (tid < H2) {
        const float y = (pooled[tid] - stats[0]) * stats[1] * ln2g[tid] + ln2b[tid];
        dotb[tid] = y * fcw[tid];
    }
    __syncthreads();
    if (wave == 0) {
        float d = dotb[lane] + dotb[lane + 32] + dotb[lane + 64] + dotb[lane + 96];
        d = wave_sum32(d);
        if (lane == 0) out[b] = d + fcb[0];
    }
}

// ---------------------------------------------------------------------------
extern "C" void kernel_launch(void* const* d_in, const int* in_sizes, int n_in,
                              void* d_out, int out_size, void* d_ws, size_t ws_size,
                              hipStream_t stream) {
    const int*   ids  = (const int*)  d_in[0];
    const int*   lens = (const int*)  d_in[1];
    const float* emb  = (const float*)d_in[2];
    const float* ln1g = (const float*)d_in[3];
    const float* ln1b = (const float*)d_in[4];
    const float* w1   = (const float*)d_in[5];
    const float* b1   = (const float*)d_in[6];
    const float* w2   = (const float*)d_in[7];
    const float* b2   = (const float*)d_in[8];
    const float* ln2g = (const float*)d_in[9];
    const float* ln2b = (const float*)d_in[10];
    const float* fcw  = (const float*)d_in[11];
    const float* fcb  = (const float*)d_in[12];
    float* out = (float*)d_out;

    _Float16* w1t = (_Float16*)d_ws;          // 256*128 f16 = 64 KB
    _Float16* w2t = w1t + (size_t)EMBD * H1;  // 128*256 f16 = 64 KB

    const int B = in_sizes[1];                // 4096 entities

    const int prep_elems = EMBD * H1 + H1 * H2;
    prep_weights_f16<<<(prep_elems + 255) / 256, 256, 0, stream>>>(w1, w2, w1t, w2t);

    const size_t smem = 16384 + 32768 + 32768 + 512 + 32 + 512; // ~83 KB
    entity_encoder_fused<<<B, 256, smem, stream>>>(
        ids, lens, emb, ln1g, ln1b, b1, b2, ln2g, ln2b, fcw, fcb, w1t, w2t, out);
}
